// Attention_687194768283
// MI455X (gfx1250) — matrix operand order repaired and checked by reading, not compiled, not run on hardware
//
#include <hip/hip_runtime.h>
#include <hip/hip_bf16.h>

// ---------------------------------------------------------------------------
// Fused temporal attention for MI455X (gfx1250), wave32 + bf16 WMMA.
//   qkv = x @ w_qkv^T   (bf16 inputs, f32 accum, v_wmma_f32_16x16x32_bf16)
//   sim = (q*scale) @ k^T + pos_bias ; focus-present mask ; softmax
//   out = attn @ v
// One workgroup per (b, s): 64 tokens, dim 512, 8 heads of 64.
// B operand (w_qkv) is staged per-head into LDS in double-buffered K-chunks
// using GLOBAL_LOAD_ASYNC_TO_LDS_B128 (ASYNCcnt), overlapping global latency
// with WMMA compute and with the softmax of the previous head.
// ---------------------------------------------------------------------------

typedef __attribute__((ext_vector_type(16))) __bf16        v16bf;
typedef __attribute__((ext_vector_type(8)))  float         v8f;
typedef __attribute__((ext_vector_type(4)))  unsigned int  uint4v;
typedef __attribute__((ext_vector_type(2)))  unsigned int  uint2v;
typedef __attribute__((ext_vector_type(4)))  float         float4v;
typedef int v4i __attribute__((__vector_size__(16)));      // matches builtin

union FragB16 { v16bf v; uint4v u[2]; };

#if defined(__has_builtin)
#if __has_builtin(__builtin_amdgcn_global_load_async_to_lds_b128) && \
    __has_builtin(__builtin_amdgcn_s_wait_asynccnt)
#define USE_ASYNC 1
#endif
#endif
#ifndef USE_ASYNC
#define USE_ASYNC 0
#endif

#define AS1 __attribute__((address_space(1)))
#define AS3 __attribute__((address_space(3)))

__device__ __forceinline__ unsigned int f2bf(float f) {
    // round-to-nearest-even f32 -> bf16 (raw u16 in low bits)
    unsigned int u = __float_as_uint(f);
    unsigned int r = ((u >> 16) & 1u) + 0x7FFFu;
    return (u + r) >> 16;
}

__device__ __forceinline__ void wait_async() {
#if USE_ASYNC
    __builtin_amdgcn_s_wait_asynccnt(0);
#endif
}

// Stage one K-chunk of B (192 rows x 32 cols bf16) for head h into LDS.
// Row r of the chunk is w_qkv row e = (r/64)*512 + h*64 + (r%64); each of the
// 768 16-byte segments is copied by one of the 256 threads (3 each).
__device__ __forceinline__ void issue_chunk(const unsigned short* __restrict__ wb,
                                            unsigned short* bbuf,
                                            int h, int kc, int t) {
    #pragma unroll
    for (int i = 0; i < 3; ++i) {
        int id  = t + i * 256;                 // 0..767
        int r   = id >> 2;                     // chunk row 0..191
        int seg = id & 3;                      // 16B segment within row
        int e   = ((r >> 6) << 9) + (r & 63);  // + h*64 below
        const unsigned short* gp = wb + (size_t)(e + h * 64) * 512 + kc * 32 + seg * 8;
        unsigned short*       lp = bbuf + r * 40 + seg * 8;
#if USE_ASYNC
        __builtin_amdgcn_global_load_async_to_lds_b128(
            (AS1 v4i*)gp, (AS3 v4i*)lp, 0, 0);
#else
        *(uint4v*)lp = *(const uint4v*)gp;
#endif
    }
}

// --------------------------- w_qkv fp32 -> bf16 ----------------------------
__global__ void wqkv_to_bf16_kernel(const float4v* __restrict__ w,
                                    uint2v* __restrict__ wb) {
    int i = blockIdx.x * blockDim.x + threadIdx.x;   // 196608 float4s
    float4v v = w[i];
    uint2v o;
    o.x = f2bf(v.x) | (f2bf(v.y) << 16);
    o.y = f2bf(v.z) | (f2bf(v.w) << 16);
    wb[i] = o;
}

// ------------------------------ main kernel --------------------------------
#define XS 520   // x_s row stride in ushorts (512 + 8 pad, keeps 16B align)
#define BS 40    // staged B-chunk row stride (32 + 8 pad)
#define TS 72    // 64x64 bf16 tile row stride
#define SS 68    // sim f32 row stride

__global__ __launch_bounds__(256)
void attn_fused_kernel(const float* __restrict__ x,
                       const unsigned short* __restrict__ wb,   // 1536x512 bf16
                       const float* __restrict__ pos_bias,      // 8x64x64
                       const unsigned char* __restrict__ focus, // (b,) bool
                       float* __restrict__ out) {
    __shared__ unsigned short x_s [64 * XS];     // x tile bf16
    __shared__ unsigned short b_s [2][192 * BS]; // double-buffered w chunk
    __shared__ unsigned short q_s [64 * TS];     // q (pre-scaled) bf16
    __shared__ unsigned short k_s [64 * TS];     // k bf16
    __shared__ unsigned short vT_s[64 * TS];     // v^T bf16
    __shared__ float          sim_s[64 * SS];    // logits f32
    __shared__ unsigned short a_s [64 * TS];     // attn bf16

    const int t     = threadIdx.x;
    const int lane  = t & 31;
    const int wv    = t >> 5;        // wave 0..7
    const int lh    = lane & 15;     // N / M position within fragment
    const int kh    = lane >> 4;     // lane half
    const int blk   = blockIdx.x;    // b*1024 + s
    const int batch = blk >> 10;
    const bool fp   = focus[batch] != 0;

    // kick off B chunk 0 for head 0 so it lands while we stage x
    issue_chunk(wb, b_s[0], 0, 0, t);

    // ---- stage x tile: 64x512 f32 -> bf16 LDS (coalesced float4 loads) ----
    {
        const float4v* x4 = (const float4v*)(x + (size_t)blk * 32768);
        #pragma unroll
        for (int i = 0; i < 32; ++i) {
            int idx = t + i * 256;           // float4 index, 128 per row
            int row = idx >> 7;
            int c4  = idx & 127;
            float4v v = x4[idx];
            uint2v o;
            o.x = f2bf(v.x) | (f2bf(v.y) << 16);
            o.y = f2bf(v.z) | (f2bf(v.w) << 16);
            *(uint2v*)&x_s[row * XS + c4 * 4] = o;
        }
    }

    const int mrow = wv & 1;         // GEMM: 2 m-tiles per wave
    const int ncol = wv >> 1;        // GEMM: 3 n-tiles per wave (0..3)
    const int mt   = wv >> 1;        // attention: m-tile 0..3
    const int half = wv & 1;         // attention: j/d half

    for (int h = 0; h < 8; ++h) {
        wait_async();
        __syncthreads();             // chunk 0 (and x_s on h==0) ready

        // ================= QKV projection: 64x192, K=512 =================
        v8f acc[6] = {};             // [mm*3 + tt]
        for (int kc = 0; kc < 16; ++kc) {
            int cur = kc & 1;
            if (kc < 15) issue_chunk(wb, b_s[cur ^ 1], h, kc + 1, t);

            FragB16 a[2], b[3];
            #pragma unroll
            for (int mm = 0; mm < 2; ++mm) {
                const unsigned short* xr =
                    &x_s[((mrow * 2 + mm) * 16 + lh) * XS + kc * 32];
                a[mm].u[0] = *(const uint4v*)(xr + kh * 8);
                a[mm].u[1] = *(const uint4v*)(xr + 16 + kh * 8);
            }
            #pragma unroll
            for (int tt = 0; tt < 3; ++tt) {
                const unsigned short* bp =
                    &b_s[cur][((ncol * 3 + tt) * 16 + lh) * BS + kh * 8];
                b[tt].u[0] = *(const uint4v*)(bp);
                b[tt].u[1] = *(const uint4v*)(bp + 16);
            }
            #pragma unroll
            for (int mm = 0; mm < 2; ++mm)
                #pragma unroll
                for (int tt = 0; tt < 3; ++tt)
                    acc[mm * 3 + tt] = __builtin_amdgcn_wmma_f32_16x16x32_bf16(
                        false, a[mm].v, false, b[tt].v, (short)0,
                        acc[mm * 3 + tt], false, false);

            wait_async();            // next chunk landed during compute
            __syncthreads();
        }
        // scatter accumulators: q scaled, k row-major, v transposed
        #pragma unroll
        for (int mm = 0; mm < 2; ++mm) {
            #pragma unroll
            for (int tt = 0; tt < 3; ++tt) {
                int nt = ncol * 3 + tt;
                #pragma unroll
                for (int r = 0; r < 8; ++r) {
                    int m = (mrow * 2 + mm) * 16 + r + kh * 8;
                    int n = nt * 16 + lh;
                    float vv = acc[mm * 3 + tt][r];
                    if (nt < 4)       q_s [m * TS + n]         = f2bf(vv * 0.125f);
                    else if (nt < 8)  k_s [m * TS + (n - 64)]  = f2bf(vv);
                    else              vT_s[(n - 128) * TS + m] = f2bf(vv);
                }
            }
        }
        // prefetch next head's first B chunk; latency hides under attention
        if (h < 7) issue_chunk(wb, b_s[0], h + 1, 0, t);
        __syncthreads();

        // ================= sim = q @ k^T (64x64, K=64) ===================
        {
            const int jb = half * 2;     // 2 j-tiles per wave
            v8f s0 = {}, s1 = {};
            #pragma unroll
            for (int kc = 0; kc < 2; ++kc) {
                FragB16 a;
                const unsigned short* qr = &q_s[(mt * 16 + lh) * TS + kc * 32];
                a.u[0] = *(const uint4v*)(qr + kh * 8);
                a.u[1] = *(const uint4v*)(qr + 16 + kh * 8);
                #pragma unroll
                for (int tt = 0; tt < 2; ++tt) {
                    const unsigned short* kp =
                        &k_s[((jb + tt) * 16 + lh) * TS + kc * 32 + kh * 8];
                    FragB16 b;
                    b.u[0] = *(const uint4v*)(kp);
                    b.u[1] = *(const uint4v*)(kp + 16);
                    v8f& s = tt ? s1 : s0;
                    s = __builtin_amdgcn_wmma_f32_16x16x32_bf16(
                        false, a.v, false, b.v, (short)0, s, false, false);
                }
            }
            #pragma unroll
            for (int tt = 0; tt < 2; ++tt) {
                v8f s = tt ? s1 : s0;
                #pragma unroll
                for (int r = 0; r < 8; ++r) {
                    int i = mt * 16 + r + kh * 8;
                    int j = (jb + tt) * 16 + lh;
                    sim_s[i * SS + j] = s[r];
                }
            }
        }
        __syncthreads();

        // ====== bias + mask + softmax (4 lanes per row, shfl reduce) =====
        {
            int row = t >> 2;
            int q4  = t & 3;
            const float4v* pb4 =
                (const float4v*)(pos_bias + h * 4096 + row * 64 + q4 * 16);
            float vals[16];
            float mx = -3.4e38f;
            #pragma unroll
            for (int e4 = 0; e4 < 4; ++e4) {
                float4v pv = pb4[e4];
                #pragma unroll
                for (int c = 0; c < 4; ++c) {
                    int col = q4 * 16 + e4 * 4 + c;
                    float sv = sim_s[row * SS + col] + pv[c];
                    if (fp && (col != row)) sv = -3.0e38f;  // eye mask
                    vals[e4 * 4 + c] = sv;
                    mx = fmaxf(mx, sv);
                }
            }
            mx = fmaxf(mx, __shfl_xor(mx, 1));
            mx = fmaxf(mx, __shfl_xor(mx, 2));
            float sum = 0.f;
            #pragma unroll
            for (int e = 0; e < 16; ++e) {
                vals[e] = __expf(vals[e] - mx);
                sum += vals[e];
            }
            sum += __shfl_xor(sum, 1);
            sum += __shfl_xor(sum, 2);
            float inv = 1.0f / sum;
            #pragma unroll
            for (int e = 0; e < 16; ++e)
                a_s[row * TS + q4 * 16 + e] = f2bf(vals[e] * inv);
        }
        __syncthreads();

        // ================= out = attn @ v (64x64, K=64) ==================
        {
            const int db = half * 2;     // 2 d-tiles per wave
            v8f o0 = {}, o1 = {};
            #pragma unroll
            for (int kc = 0; kc < 2; ++kc) {
                FragB16 a;
                const unsigned short* ar = &a_s[(mt * 16 + lh) * TS + kc * 32];
                a.u[0] = *(const uint4v*)(ar + kh * 8);
                a.u[1] = *(const uint4v*)(ar + 16 + kh * 8);
                #pragma unroll
                for (int tt = 0; tt < 2; ++tt) {
                    const unsigned short* vp =
                        &vT_s[((db + tt) * 16 + lh) * TS + kc * 32 + kh * 8];
                    FragB16 b;
                    b.u[0] = *(const uint4v*)(vp);
                    b.u[1] = *(const uint4v*)(vp + 16);
                    v8f& o = tt ? o1 : o0;
                    o = __builtin_amdgcn_wmma_f32_16x16x32_bf16(
                        false, a.v, false, b.v, (short)0, o, false, false);
                }
            }
            float* ob = out + (size_t)blk * 32768 + h * 64;
            #pragma unroll
            for (int tt = 0; tt < 2; ++tt) {
                v8f o = tt ? o1 : o0;
                #pragma unroll
                for (int r = 0; r < 8; ++r) {
                    int m = mt * 16 + r + kh * 8;       // token
                    int n = (db + tt) * 16 + lh;        // head dim
                    ob[m * 512 + n] = o[r];
                }
            }
        }
        __syncthreads();   // LDS reused next head
    }
}

// ------------------------------- launcher ----------------------------------
extern "C" void kernel_launch(void* const* d_in, const int* in_sizes, int n_in,
                              void* d_out, int out_size, void* d_ws, size_t ws_size,
                              hipStream_t stream) {
    const float*         x  = (const float*)d_in[0];          // (2,1024,64,512)
    const float*         w  = (const float*)d_in[1];          // (1536,512)
    const float*         pb = (const float*)d_in[2];          // (8,64,64)
    const unsigned char* fm = (const unsigned char*)d_in[3];  // (2,) bool
    float*               o  = (float*)d_out;                  // (2,1024,64,512)
    unsigned short*      wbf = (unsigned short*)d_ws;         // bf16 w_qkv

    // 1536*512 / 4 = 196608 float4s -> 768 blocks
    wqkv_to_bf16_kernel<<<768, 256, 0, stream>>>((const float4v*)w, (uint2v*)wbf);
    // one workgroup per (b, s)
    attn_fused_kernel<<<2048, 256, 0, stream>>>(x, wbf, pb, fm, o);
}
